// TransformerBlock_43456479101842
// MI455X (gfx1250) — matrix operand order repaired and checked by reading, not compiled, not run on hardware
//
#include <hip/hip_runtime.h>
#include <hip/hip_bf16.h>

// ---------------- problem constants ----------------
#define BB    4
#define TT    2048
#define CC    1024
#define NHEAD 16
#define HDIM  64
#define DFF   4096
#define MM    (BB*TT)      // 8192 rows
#define NQKV  (3*CC)       // 3072

typedef __attribute__((ext_vector_type(16))) __bf16 bh16;
typedef __attribute__((ext_vector_type(8)))  float  f32x8;

union FragU { bh16 v; uint4 u[2]; };

__device__ __forceinline__ f32x8 wmma_bf16(const FragU& a, const FragU& b, f32x8 c) {
  return __builtin_amdgcn_wmma_f32_16x16x32_bf16(false, a.v, false, b.v,
                                                 (short)0, c, false, false);
}

// async global->LDS copy (gfx1250), GV mode: 64-bit vaddr, 32-bit LDS dest addr
__device__ __forceinline__ void async_copy_b128(const void* gptr, void* lptr) {
  unsigned lds = (unsigned)(unsigned long long)lptr;   // low 32 bits == LDS offset
  asm volatile("global_load_async_to_lds_b128 %0, %1, off"
               :: "v"(lds), "v"(gptr) : "memory");
}
__device__ __forceinline__ void async_wait0() {
  asm volatile("s_wait_asynccnt 0x0" ::: "memory");
}

// ---------------- LayerNorm: fp32 in -> bf16 out ----------------
__global__ __launch_bounds__(256) void k_layernorm_bf16(
    const float* __restrict__ x, const float* __restrict__ g,
    const float* __restrict__ bta, __hip_bfloat16* __restrict__ out, int C)
{
  const int row  = blockIdx.x;
  const int tid  = threadIdx.x;
  const int lane = tid & 31, wid = tid >> 5;
  const float* xr = x + (size_t)row * C;

  float vals[4];
  float s = 0.f, ss = 0.f;
#pragma unroll
  for (int k = 0; k < 4; ++k) {
    float v = xr[tid + k * 256];
    vals[k] = v; s += v; ss += v * v;
  }
#pragma unroll
  for (int off = 16; off >= 1; off >>= 1) {
    s  += __shfl_xor(s,  off);
    ss += __shfl_xor(ss, off);
  }
  __shared__ float sbuf[8], ssbuf[8];
  if (lane == 0) { sbuf[wid] = s; ssbuf[wid] = ss; }
  __syncthreads();
  float ts = 0.f, tss = 0.f;
#pragma unroll
  for (int w = 0; w < 8; ++w) { ts += sbuf[w]; tss += ssbuf[w]; }
  const float inv = 1.0f / (float)C;
  const float mu  = ts * inv;
  const float var = tss * inv - mu * mu;
  const float rstd = rsqrtf(var + 1e-5f);

  __hip_bfloat16* orow = out + (size_t)row * C;
#pragma unroll
  for (int k = 0; k < 4; ++k) {
    int c = tid + k * 256;
    orow[c] = __float2bfloat16((vals[k] - mu) * rstd * g[c] + bta[c]);
  }
}

// ---------------- transpose + fp32->bf16: W[K x N] -> Wt[N x K] ----------------
__global__ __launch_bounds__(256) void k_transpose_bf16(
    const float* __restrict__ w, __hip_bfloat16* __restrict__ wt, int K, int N)
{
  int idx = blockIdx.x * 256 + threadIdx.x;
  if (idx >= K * N) return;
  int k = idx / N, n = idx - k * N;
  wt[(size_t)n * K + k] = __float2bfloat16(w[idx]);
}

// ---------------- WMMA GEMM: out = A[MxK](bf16) * Bt[NxK](bf16)^T + epi ----------------
#define EPI_BF16          0   // write bf16
#define EPI_RES_F32       1   // +res, write f32
#define EPI_BIAS_GELU_B16 2   // +bias, exact gelu, write bf16
#define EPI_BIAS_RES_F32  3   // +bias +res, write f32

// block tile 128x128, 8 waves (4 row-groups x 2 col-groups), wave tile 32x64
template<int EPI>
__global__ __launch_bounds__(256) void k_gemm_bf16(
    const __hip_bfloat16* __restrict__ A,  // M x K
    const __hip_bfloat16* __restrict__ Bt, // N x K
    const float* __restrict__ bias,
    const float* __restrict__ res,
    float* __restrict__ outF, __hip_bfloat16* __restrict__ outH,
    int M, int N, int K)
{
  const int lane = threadIdx.x & 31, wid = threadIdx.x >> 5;
  const int hf   = lane >> 4;
  const int i16  = lane & 15;
  const int row0 = blockIdx.y * 128 + (wid >> 1) * 32;
  const int col0 = blockIdx.x * 128 + (wid & 1)  * 64;

  const __hip_bfloat16* aptr[2];
  const __hip_bfloat16* bptr[4];
#pragma unroll
  for (int i = 0; i < 2; ++i) aptr[i] = A  + (size_t)(row0 + i * 16 + i16) * K;
#pragma unroll
  for (int j = 0; j < 4; ++j) bptr[j] = Bt + (size_t)(col0 + j * 16 + i16) * K;

  f32x8 acc[2][4] = {};

#pragma unroll 2
  for (int k = 0; k < K; k += 32) {
    FragU a[2], b[4];
#pragma unroll
    for (int i = 0; i < 2; ++i) {
      const __hip_bfloat16* ap = aptr[i] + k + hf * 8;
      a[i].u[0] = *(const uint4*)ap;
      a[i].u[1] = *(const uint4*)(ap + 16);
    }
#pragma unroll
    for (int j = 0; j < 4; ++j) {
      const __hip_bfloat16* bp = bptr[j] + k + hf * 8;
      b[j].u[0] = *(const uint4*)bp;
      b[j].u[1] = *(const uint4*)(bp + 16);
    }
#pragma unroll
    for (int i = 0; i < 2; ++i)
#pragma unroll
      for (int j = 0; j < 4; ++j)
        acc[i][j] = wmma_bf16(a[i], b[j], acc[i][j]);
  }

#pragma unroll
  for (int i = 0; i < 2; ++i) {
#pragma unroll
    for (int j = 0; j < 4; ++j) {
      const int mbase = row0 + i * 16 + hf * 8;
      const int n     = col0 + j * 16 + i16;
#pragma unroll
      for (int r = 0; r < 8; ++r) {
        float v = acc[i][j][r];
        size_t off = (size_t)(mbase + r) * N + n;
        if (EPI == EPI_BF16) {
          outH[off] = __float2bfloat16(v);
        } else if (EPI == EPI_RES_F32) {
          outF[off] = v + res[off];
        } else if (EPI == EPI_BIAS_GELU_B16) {
          v += bias[n];
          v = 0.5f * v * (1.0f + erff(v * 0.70710678118654752f));
          outH[off] = __float2bfloat16(v);
        } else { // EPI_BIAS_RES_F32
          outF[off] = v + bias[n] + res[off];
        }
      }
    }
  }
}

// ---------------- fused flash attention (bf16 QKV -> bf16 out) ----------------
// qkv: [B, T, 3C] bf16 (row stride 3072). out: [B, T, C] bf16.
// 256 threads = 8 waves; each wave owns 16 query rows (128 per block); key blocks of 64.
__global__ __launch_bounds__(256) void k_flash_attn(
    const __hip_bfloat16* __restrict__ qkv, __hip_bfloat16* __restrict__ out)
{
  const int S = NQKV;
  const int lane = threadIdx.x & 31, w = threadIdx.x >> 5;
  const int hf = lane >> 4, i16 = lane & 15;
  const int h = blockIdx.y, b = blockIdx.z;
  const int qm0 = blockIdx.x * 128 + w * 16;

  const __hip_bfloat16* Q  = qkv + (size_t)b * TT * S + h * HDIM;
  const __hip_bfloat16* Kp = Q + CC;
  const __hip_bfloat16* Vp = Q + 2 * CC;

  // Q fragments resident in VGPRs: 16 rows x 64 K (two k-steps of 32)
  FragU aq[2];
#pragma unroll
  for (int kk = 0; kk < 2; ++kk) {
    const __hip_bfloat16* qp = Q + (size_t)(qm0 + i16) * S + kk * 32 + hf * 8;
    aq[kk].u[0] = *(const uint4*)qp;
    aq[kk].u[1] = *(const uint4*)(qp + 16);
  }

  __shared__ __hip_bfloat16 Klds[64][64];     // key block (t x d), row-major  (8KB)
  __shared__ __hip_bfloat16 Vtlds[64][64];    // value block transposed (d x t) (8KB)
  __shared__ __hip_bfloat16 Plds[8][16][64];  // per-wave P staging            (16KB)

  float m_i[8], l_i[8];
#pragma unroll
  for (int r = 0; r < 8; ++r) { m_i[r] = -1e30f; l_i[r] = 0.f; }
  f32x8 accO[4] = {};

  // staging geometry (256 threads over 64x64 tiles)
  const int srow = threadIdx.x >> 2;          // 0..63
  const int scg  = (threadIdx.x & 3) * 16;    // 0,16,32,48

  for (int kb = 0; kb < TT; kb += 64) {
    __syncthreads();

    // ---- async-stage K block into LDS (CDNA5 async copy path) ----
    {
      const __hip_bfloat16* krow = Kp + (size_t)(kb + srow) * S + scg;
      async_copy_b128(krow,     &Klds[srow][scg]);
      async_copy_b128(krow + 8, &Klds[srow][scg + 8]);
    }
    // ---- stage V transposed (through VGPRs) ----
    {
      const __hip_bfloat16* vrow = Vp + (size_t)(kb + srow) * S + scg;
      uint4 v0 = *(const uint4*)vrow;
      uint4 v1 = *(const uint4*)(vrow + 8);
      const __hip_bfloat16* e0 = (const __hip_bfloat16*)&v0;
      const __hip_bfloat16* e1 = (const __hip_bfloat16*)&v1;
#pragma unroll
      for (int c = 0; c < 8; ++c) Vtlds[scg + c][srow]     = e0[c];
#pragma unroll
      for (int c = 0; c < 8; ++c) Vtlds[scg + 8 + c][srow] = e1[c];
    }
    async_wait0();
    __syncthreads();

    // ---- S = Q K^T : 16 query rows x 64 keys (4 n-tiles) ----
    f32x8 accS[4] = {};
#pragma unroll
    for (int nj = 0; nj < 4; ++nj) {
#pragma unroll
      for (int kk = 0; kk < 2; ++kk) {
        FragU bk;
        const __hip_bfloat16* kp = &Klds[nj * 16 + i16][kk * 32 + hf * 8];
        bk.u[0] = *(const uint4*)kp;
        bk.u[1] = *(const uint4*)(kp + 16);
        accS[nj] = wmma_bf16(aq[kk], bk, accS[nj]);
      }
    }

    // ---- online softmax (row = hf*8 + r, col = i16) ----
    float p[4][8];
#pragma unroll
    for (int nj = 0; nj < 4; ++nj)
#pragma unroll
      for (int r = 0; r < 8; ++r) p[nj][r] = accS[nj][r] * 0.125f;  // 1/sqrt(64)

#pragma unroll
    for (int r = 0; r < 8; ++r) {
      float mx = fmaxf(fmaxf(p[0][r], p[1][r]), fmaxf(p[2][r], p[3][r]));
      mx = fmaxf(mx, __shfl_xor(mx, 1));
      mx = fmaxf(mx, __shfl_xor(mx, 2));
      mx = fmaxf(mx, __shfl_xor(mx, 4));
      mx = fmaxf(mx, __shfl_xor(mx, 8));
      float mn   = fmaxf(m_i[r], mx);
      float corr = __expf(m_i[r] - mn);
      m_i[r] = mn;
      float rs = 0.f;
#pragma unroll
      for (int nj = 0; nj < 4; ++nj) {
        p[nj][r] = __expf(p[nj][r] - mn);
        rs += p[nj][r];
      }
      rs += __shfl_xor(rs, 1);
      rs += __shfl_xor(rs, 2);
      rs += __shfl_xor(rs, 4);
      rs += __shfl_xor(rs, 8);
      l_i[r] = l_i[r] * corr + rs;
#pragma unroll
      for (int j = 0; j < 4; ++j) accO[j][r] *= corr;
    }

    // ---- P: C-layout -> LDS -> A-layout (per-wave tile; DS in-order) ----
#pragma unroll
    for (int nj = 0; nj < 4; ++nj)
#pragma unroll
      for (int r = 0; r < 8; ++r)
        Plds[w][hf * 8 + r][nj * 16 + i16] = __float2bfloat16(p[nj][r]);

    // ---- O += P * V  (two k-steps of 32 over the 64 keys) ----
#pragma unroll
    for (int kk2 = 0; kk2 < 2; ++kk2) {
      FragU pa;
      const __hip_bfloat16* pp = &Plds[w][i16][kk2 * 32 + hf * 8];
      pa.u[0] = *(const uint4*)pp;
      pa.u[1] = *(const uint4*)(pp + 16);
#pragma unroll
      for (int j = 0; j < 4; ++j) {
        FragU bv;
        const __hip_bfloat16* vp = &Vtlds[j * 16 + i16][kk2 * 32 + hf * 8];
        bv.u[0] = *(const uint4*)vp;
        bv.u[1] = *(const uint4*)(vp + 16);
        accO[j] = wmma_bf16(pa, bv, accO[j]);
      }
    }
  }

  // ---- write O / l as bf16 into [B,T,C] ----
#pragma unroll
  for (int j = 0; j < 4; ++j) {
#pragma unroll
    for (int r = 0; r < 8; ++r) {
      float v = accO[j][r] / l_i[r];
      int mg = qm0 + hf * 8 + r;
      out[((size_t)b * TT + mg) * CC + h * HDIM + j * 16 + i16] = __float2bfloat16(v);
    }
  }
}

// ---------------- host-side orchestration ----------------
extern "C" void kernel_launch(void* const* d_in, const int* in_sizes, int n_in,
                              void* d_out, int out_size, void* d_ws, size_t ws_size,
                              hipStream_t stream)
{
  (void)in_sizes; (void)n_in; (void)out_size; (void)ws_size;
  const float* x     = (const float*)d_in[0];
  const float* ln1_g = (const float*)d_in[1];
  const float* ln1_b = (const float*)d_in[2];
  const float* ln2_g = (const float*)d_in[3];
  const float* ln2_b = (const float*)d_in[4];
  const float* w_qkv = (const float*)d_in[5];
  const float* w_prj = (const float*)d_in[6];
  const float* w1    = (const float*)d_in[7];
  const float* b1    = (const float*)d_in[8];
  const float* w2    = (const float*)d_in[9];
  const float* b2    = (const float*)d_in[10];
  float* out = (float*)d_out;

  char* ws = (char*)d_ws;
  size_t off = 0;
  auto alloc = [&](size_t bytes) { size_t o = off; off += (bytes + 255) & ~(size_t)255; return o; };

  __hip_bfloat16* xn     = (__hip_bfloat16*)(ws + alloc((size_t)MM * CC   * 2));
  __hip_bfloat16* wqkvT  = (__hip_bfloat16*)(ws + alloc((size_t)NQKV * CC * 2));
  __hip_bfloat16* wprjT  = (__hip_bfloat16*)(ws + alloc((size_t)CC * CC   * 2));
  __hip_bfloat16* w1T    = (__hip_bfloat16*)(ws + alloc((size_t)DFF * CC  * 2));
  __hip_bfloat16* w2T    = (__hip_bfloat16*)(ws + alloc((size_t)CC * DFF  * 2));
  __hip_bfloat16* qkv    = (__hip_bfloat16*)(ws + alloc((size_t)MM * NQKV * 2));
  __hip_bfloat16* attn   = (__hip_bfloat16*)(ws + alloc((size_t)MM * CC   * 2));
  float*          xmid   = (float*)         (ws + alloc((size_t)MM * CC   * 4));
  __hip_bfloat16* hg     = (__hip_bfloat16*)(ws + alloc((size_t)MM * DFF  * 2));

  // 1. LN1(x) -> xn (bf16)
  k_layernorm_bf16<<<MM, 256, 0, stream>>>(x, ln1_g, ln1_b, xn, CC);

  // 2. transpose weights to bf16 [N x K]
  k_transpose_bf16<<<(CC * NQKV + 255) / 256, 256, 0, stream>>>(w_qkv, wqkvT, CC, NQKV);
  k_transpose_bf16<<<(CC * CC   + 255) / 256, 256, 0, stream>>>(w_prj, wprjT, CC, CC);
  k_transpose_bf16<<<(CC * DFF  + 255) / 256, 256, 0, stream>>>(w1,    w1T,   CC, DFF);
  k_transpose_bf16<<<(DFF * CC  + 255) / 256, 256, 0, stream>>>(w2,    w2T,   DFF, CC);

  // 3. qkv = xn @ w_qkv   (bf16 out)
  k_gemm_bf16<EPI_BF16><<<dim3(NQKV / 128, MM / 128), 256, 0, stream>>>(
      xn, wqkvT, nullptr, nullptr, nullptr, qkv, MM, NQKV, CC);

  // 4. fused flash attention -> attn (bf16)
  k_flash_attn<<<dim3(TT / 128, NHEAD, BB), 256, 0, stream>>>(qkv, attn);

  // 5. xmid = x + attn @ w_proj  (f32)
  k_gemm_bf16<EPI_RES_F32><<<dim3(CC / 128, MM / 128), 256, 0, stream>>>(
      attn, wprjT, nullptr, x, xmid, nullptr, MM, CC, CC);

  // 6. LN2(xmid) -> xn (reuse, bf16)
  k_layernorm_bf16<<<MM, 256, 0, stream>>>(xmid, ln2_g, ln2_b, xn, CC);

  // 7. hg = gelu(xn @ w1 + b1)  (bf16)
  k_gemm_bf16<EPI_BIAS_GELU_B16><<<dim3(DFF / 128, MM / 128), 256, 0, stream>>>(
      xn, w1T, b1, nullptr, nullptr, hg, MM, DFF, CC);

  // 8. out = xmid + hg @ w2 + b2  (f32)
  k_gemm_bf16<EPI_BIAS_RES_F32><<<dim3(CC / 128, MM / 128), 256, 0, stream>>>(
      hg, w2T, b2, xmid, out, nullptr, MM, CC, DFF);
}
